// DistSAGEModel_75582834475277
// MI455X (gfx1250) — matrix operand order repaired and checked by reading, not compiled, not run on hardware
//
#include <hip/hip_runtime.h>

// ---------------------------------------------------------------------------
// GraphSAGE 3-layer inference for MI455X (gfx1250, wave32, WMMA).
//   h = relu( X[:n_dst] @ Wself + segsum(X[src]->dst) @ Wneigh + b )
// Scatter-aggregate is HBM-bound (~3.5 GB traffic layer 0); GEMMs use
// v_wmma_f32_16x16x32_bf16 with f32 accumulation.
// ---------------------------------------------------------------------------

typedef __attribute__((ext_vector_type(16))) __bf16 v16bf;
typedef __attribute__((ext_vector_type(8)))  float  v8f;

#define D_HALF   256   // K of each half (self / neigh); total K = 512
#define KT_TOTAL 16    // 512 / 32 k-steps

// ---------------------------------------------------------------- zero fill
__global__ void zero_f32_kernel(float* __restrict__ p, long long n) {
  long long i = (long long)blockIdx.x * blockDim.x + threadIdx.x;
  long long stride = (long long)gridDim.x * blockDim.x;
  for (; i < n; i += stride) p[i] = 0.0f;
}

// ------------------------------------------------- pack weights to B-frags
// Wfull[k][n] = k<256 ? ws[k][n] : wn[k-256][n]   (row-major, ld = Nout)
// Fragment (nt,kt): 32 lanes x 16 bf16, lane l -> col n = nt*16+(l&15),
// K = kt*32 + 8*(l>>4) + (e<8 ? e : 8+e)  (ISA 16-bit table order).
__global__ void pack_w_kernel(const float* __restrict__ ws,
                              const float* __restrict__ wn,
                              __bf16* __restrict__ wp, int Nout, int NT) {
  int idx = blockIdx.x * blockDim.x + threadIdx.x;
  int total = NT * KT_TOTAL * 512;
  if (idx >= total) return;
  int e    = idx & 15;
  int lane = (idx >> 4) & 31;
  int kt   = (idx >> 9) & 15;
  int nt   = idx >> 13;
  int n = nt * 16 + (lane & 15);
  int k = kt * 32 + (lane >> 4) * 8 + (e < 8 ? e : 8 + e);
  float v = 0.0f;
  if (n < Nout)
    v = (k < D_HALF) ? ws[(long long)k * Nout + n]
                     : wn[(long long)(k - D_HALF) * Nout + n];
  wp[idx] = (__bf16)v;
}

// ------------------------------------------------- edge scatter-aggregate
// One wave per edge: lane reads float4 pairs of x[src], f32 atomic-add into
// agg[dst] (global_atomic_add_f32, relaxed, device scope).
__global__ void scatter_agg_kernel(const float* __restrict__ x,
                                   const int* __restrict__ src,
                                   const int* __restrict__ dst,
                                   float* __restrict__ agg, int nE) {
  long long wid = (long long)blockIdx.x * (blockDim.x >> 5) + (threadIdx.x >> 5);
  if (wid >= nE) return;
  int lane = threadIdx.x & 31;
  int s = src[wid];
  int d = dst[wid];
  const float4* xs = (const float4*)(x + (long long)s * D_HALF) + lane;
  float* ad = agg + (long long)d * D_HALF + lane * 4;
#pragma unroll
  for (int h = 0; h < 2; ++h) {
    float4 v = xs[h * 32];
    float* p = ad + h * 128;
    __hip_atomic_fetch_add(p + 0, v.x, __ATOMIC_RELAXED, __HIP_MEMORY_SCOPE_AGENT);
    __hip_atomic_fetch_add(p + 1, v.y, __ATOMIC_RELAXED, __HIP_MEMORY_SCOPE_AGENT);
    __hip_atomic_fetch_add(p + 2, v.z, __ATOMIC_RELAXED, __HIP_MEMORY_SCOPE_AGENT);
    __hip_atomic_fetch_add(p + 3, v.w, __ATOMIC_RELAXED, __HIP_MEMORY_SCOPE_AGENT);
  }
}

// ------------------------------------------------------------- WMMA GEMM
// out[m,n] = act( sum_k bf16(A[m,k]) * bf16(W[k,n]) + b[n] )
// A = [aSelf | aNeigh] (f32, each ld=256). Block = 16 output rows;
// each wave owns TPW contiguous 16-wide n-tiles.
template <int TPW, bool RELU>
__global__ void sage_gemm_kernel(const float* __restrict__ aSelf,
                                 const float* __restrict__ aNeigh,
                                 const __bf16* __restrict__ wp,
                                 const float* __restrict__ bias,
                                 float* __restrict__ out,
                                 int M, int Nout) {
  const int lane = threadIdx.x & 31;
  const int wave = threadIdx.x >> 5;
  const int m0 = blockIdx.x * 16;
  int mrow = m0 + (lane & 15);
  if (mrow >= M) mrow = M - 1;              // clamp: keeps EXEC all-ones
  const int koff = (lane >> 4) * 8;          // ISA 16-bit A-fragment layout
  const int ntBase = wave * TPW;

  v8f acc[TPW] = {};

#pragma unroll 4
  for (int kt = 0; kt < KT_TOTAL; ++kt) {
    const float* __restrict__ srcA = (kt < 8) ? aSelf : aNeigh;
    const int kbase = (kt & 7) * 32;
    const float* ar = srcA + (long long)mrow * D_HALF + kbase + koff;
    const float4* ap = (const float4*)ar;    // 32B-aligned
    float4 f0 = ap[0], f1 = ap[1];           // K: kbase+koff .. +7
    float4 f2 = ap[4], f3 = ap[5];           // K: kbase+koff+16 .. +23
    v16bf a;
    a[0]  = (__bf16)f0.x; a[1]  = (__bf16)f0.y; a[2]  = (__bf16)f0.z; a[3]  = (__bf16)f0.w;
    a[4]  = (__bf16)f1.x; a[5]  = (__bf16)f1.y; a[6]  = (__bf16)f1.z; a[7]  = (__bf16)f1.w;
    a[8]  = (__bf16)f2.x; a[9]  = (__bf16)f2.y; a[10] = (__bf16)f2.z; a[11] = (__bf16)f2.w;
    a[12] = (__bf16)f3.x; a[13] = (__bf16)f3.y; a[14] = (__bf16)f3.z; a[15] = (__bf16)f3.w;
#pragma unroll
    for (int t = 0; t < TPW; ++t) {
      const int nt = ntBase + t;
      const v16bf b =
          *(const v16bf*)(wp + ((long long)(nt * KT_TOTAL + kt) << 9) + lane * 16);
      acc[t] = __builtin_amdgcn_wmma_f32_16x16x32_bf16(
          false, a, false, b, (short)0, acc[t], false, false);
    }
  }

  // ------------------------------------------------------------- epilogue
  // C/D layout: VGPR r -> row m0 + r + 8*(lane>=16); lane&15 -> column.
  const int rbase = m0 + (lane >> 4) * 8;
  const bool fullM = (m0 + 16 <= M);         // block-uniform -> scalar branch

#pragma unroll
  for (int t = 0; t < TPW; ++t) {
    const int n = (ntBase + t) * 16 + (lane & 15);
    if (n >= Nout) continue;                 // only masks layer-2 pad columns
    const float bv = bias[n];
    float* op = out + (long long)rbase * Nout + n;
    if (fullM) {
#pragma unroll
      for (int r = 0; r < 8; ++r) {
        float v = acc[t][r] + bv;
        if (RELU) v = fmaxf(v, 0.0f);
        op[(long long)r * Nout] = v;
      }
    } else {
#pragma unroll
      for (int r = 0; r < 8; ++r) {
        if (rbase + r < M) {
          float v = acc[t][r] + bv;
          if (RELU) v = fmaxf(v, 0.0f);
          op[(long long)r * Nout] = v;
        }
      }
    }
  }
}

// ---------------------------------------------------------------- launcher
extern "C" void kernel_launch(void* const* d_in, const int* in_sizes, int n_in,
                              void* d_out, int out_size, void* d_ws, size_t ws_size,
                              hipStream_t stream) {
  (void)n_in; (void)out_size; (void)ws_size;

  const float* x        = (const float*)d_in[0];
  const int*   src0     = (const int*)d_in[1];
  const int*   dst0     = (const int*)d_in[2];
  const int*   src1     = (const int*)d_in[3];
  const int*   dst1     = (const int*)d_in[4];
  const int*   src2     = (const int*)d_in[5];
  const int*   dst2     = (const int*)d_in[6];
  const float* w_self0  = (const float*)d_in[7];
  const float* w_neigh0 = (const float*)d_in[8];
  const float* b0       = (const float*)d_in[9];
  const float* w_self1  = (const float*)d_in[10];
  const float* w_neigh1 = (const float*)d_in[11];
  const float* b1       = (const float*)d_in[12];
  const float* w_self2  = (const float*)d_in[13];
  const float* w_neigh2 = (const float*)d_in[14];
  const float* b2       = (const float*)d_in[15];

  const int E0 = in_sizes[1], E1 = in_sizes[3], E2 = in_sizes[5];
  const int N1 = 100000, N2 = 25000, N3 = 6250;   // n_dst0/1/2 (fixed problem)
  const int C  = in_sizes[15];                    // 47

  // ---- workspace carve-out (256B aligned) ----
  char* ws = (char*)d_ws;
  size_t off = 0;
  auto carve = [&](size_t bytes) -> void* {
    void* p = ws + off;
    off = (off + bytes + 255) & ~(size_t)255;
    return p;
  };
  float*  agg0 = (float*)carve((size_t)N1 * D_HALF * 4);
  float*  h1   = (float*)carve((size_t)N1 * D_HALF * 4);
  float*  agg1 = (float*)carve((size_t)N2 * D_HALF * 4);
  float*  h2   = (float*)carve((size_t)N2 * D_HALF * 4);
  float*  agg2 = (float*)carve((size_t)N3 * D_HALF * 4);
  __bf16* wp0  = (__bf16*)carve((size_t)16 * KT_TOTAL * 512 * 2);
  __bf16* wp1  = (__bf16*)carve((size_t)16 * KT_TOTAL * 512 * 2);
  __bf16* wp2  = (__bf16*)carve((size_t)3  * KT_TOTAL * 512 * 2);

  // ---- pack weights into B-fragment layout ----
  pack_w_kernel<<<(16 * 8192 + 255) / 256, 256, 0, stream>>>(w_self0, w_neigh0, wp0, 256, 16);
  pack_w_kernel<<<(16 * 8192 + 255) / 256, 256, 0, stream>>>(w_self1, w_neigh1, wp1, 256, 16);
  pack_w_kernel<<<(3  * 8192 + 255) / 256, 256, 0, stream>>>(w_self2, w_neigh2, wp2, C, 3);

  // ---- layer 0 ----
  zero_f32_kernel<<<2048, 256, 0, stream>>>(agg0, (long long)N1 * D_HALF);
  scatter_agg_kernel<<<(E0 + 7) / 8, 256, 0, stream>>>(x, src0, dst0, agg0, E0);
  sage_gemm_kernel<4, true><<<N1 / 16, 128, 0, stream>>>(x, agg0, wp0, b0, h1, N1, 256);

  // ---- layer 1 ----
  zero_f32_kernel<<<2048, 256, 0, stream>>>(agg1, (long long)N2 * D_HALF);
  scatter_agg_kernel<<<(E1 + 7) / 8, 256, 0, stream>>>(h1, src1, dst1, agg1, E1);
  sage_gemm_kernel<4, true><<<(N2 + 15) / 16, 128, 0, stream>>>(h1, agg1, wp1, b1, h2, N2, 256);

  // ---- layer 2 (no relu, N=47 via 3 padded n-tiles, 3 waves/block) ----
  zero_f32_kernel<<<512, 256, 0, stream>>>(agg2, (long long)N3 * D_HALF);
  scatter_agg_kernel<<<(E2 + 7) / 8, 256, 0, stream>>>(h2, src2, dst2, agg2, E2);
  sage_gemm_kernel<1, false><<<(N3 + 15) / 16, 96, 0, stream>>>(h2, agg2, wp2, b2,
                                                                (float*)d_out, N3, C);
}
